// LSTM_42932493091173
// MI455X (gfx1250) — matrix operand order repaired and checked by reading, not compiled, not run on hardware
//
#include <hip/hip_runtime.h>

// ---------------------------------------------------------------------------
// LSTM forward for MI455X (gfx1250, wave32, WMMA, async-to-LDS staging).
// T=512, B=128, E=H=512.  bf16 WMMA with f32 accumulation.
// ---------------------------------------------------------------------------

#define EMBED   512
#define HIDDEN  512
#define G4      2048   // 4*HIDDEN, torch gate order [i, f, g, o]
#define TSTEPS  512
#define BATCH   128

typedef __attribute__((ext_vector_type(16))) __bf16 v16bf;
typedef __attribute__((ext_vector_type(8)))  __bf16 v8bf;
typedef __attribute__((ext_vector_type(8)))  float  v8f;
typedef __attribute__((ext_vector_type(4)))  int    v4i;

#define LDS_AS __attribute__((address_space(3)))
#define GLB_AS __attribute__((address_space(1)))

// ---- async global->LDS copy (ASYNCcnt-tracked), 16 bytes per call ----------
__device__ __forceinline__ void async_copy_b128(void* lds_dst, const void* gsrc) {
#if defined(__HIP_DEVICE_COMPILE__)
#if __has_builtin(__builtin_amdgcn_global_load_async_to_lds_b128)
    __builtin_amdgcn_global_load_async_to_lds_b128(
        (GLB_AS v4i*)gsrc, (LDS_AS v4i*)lds_dst, 0, 0);
#else
    unsigned loff = (unsigned)(LDS_AS char*)lds_dst;
    asm volatile("global_load_async_to_lds_b128 %0, %1, off"
                 :: "v"(loff), "v"((unsigned long long)gsrc)
                 : "memory");
#endif
#endif
}

#if defined(__HIP_DEVICE_COMPILE__)
#if __has_builtin(__builtin_amdgcn_s_wait_asynccnt)
#define WAIT_ASYNC(n) __builtin_amdgcn_s_wait_asynccnt(n)
#else
#define WAIT_ASYNC(n) asm volatile("s_wait_asynccnt %0" :: "i"(n) : "memory")
#endif
#else
#define WAIT_ASYNC(n) ((void)0)
#endif

// ---- WMMA fragment loaders (wave32, 16-bit element layouts per CDNA5 ISA) --
// A 16x32 (MxK): lane L(0..15) row M=L, K={kb..kb+7, 16+kb..+7}, kb=8*(lane>=16).
__device__ __forceinline__ v16bf load_frag_a(const __bf16* base, int stride, int lane) {
    int m  = lane & 15;
    int kb = (lane >> 4) << 3;
    const __bf16* p = base + m * stride + kb;
    v8bf lo = *(const v8bf*)(p);
    v8bf hi = *(const v8bf*)(p + 16);
    return __builtin_shufflevector(lo, hi, 0,1,2,3,4,5,6,7,8,9,10,11,12,13,14,15);
}

// B 32x16 (KxN) stored row-major [n][k]: lane n=lane&15, K={kb..kb+7, 16+kb..+7}, kb=8*(lane>=16).
__device__ __forceinline__ v16bf load_frag_b(const __bf16* base, int stride, int lane) {
    int n  = lane & 15;
    int kb = (lane >> 4) << 3;
    const __bf16* p = base + n * stride + kb;
    v8bf lo = *(const v8bf*)(p);
    v8bf hi = *(const v8bf*)(p + 16);
    return __builtin_shufflevector(lo, hi, 0,1,2,3,4,5,6,7,8,9,10,11,12,13,14,15);
}

__device__ __forceinline__ float sigmoidf_fast(float x) {
    return 1.0f / (1.0f + __expf(-x));
}

// ---------------------------------------------------------------------------
// Kernel 0a: fold biases, downconvert weights to bf16 (L2-resident afterwards)
// ---------------------------------------------------------------------------
__global__ __launch_bounds__(256) void cvt_weights(
    const float* __restrict__ w_ih, const float* __restrict__ w_hh,
    const float* __restrict__ b_ih, const float* __restrict__ b_hh,
    __bf16* __restrict__ w_ih_bf, __bf16* __restrict__ w_hh_bf,
    float* __restrict__ bias) {
    int idx = blockIdx.x * 256 + threadIdx.x;
    if (idx < G4 * EMBED) {
        w_ih_bf[idx] = (__bf16)w_ih[idx];
        w_hh_bf[idx] = (__bf16)w_hh[idx];
    }
    if (idx < G4) bias[idx] = b_ih[idx] + b_hh[idx];
}

// ---------------------------------------------------------------------------
// Kernel 0b: downconvert X to bf16 so GEMM tile staging is a pure async copy.
// ---------------------------------------------------------------------------
__global__ __launch_bounds__(256) void cvt_x(
    const float* __restrict__ x, __bf16* __restrict__ xbf) {
    size_t i = ((size_t)blockIdx.x * 256 + threadIdx.x) * 8;
    #pragma unroll
    for (int j = 0; j < 8; ++j) xbf[i + j] = (__bf16)x[i + j];
}

// ---------------------------------------------------------------------------
// Kernel 1: input projection  xg = X_bf16 . W_ih^T + bias
// M=65536, N=2048, K=512.  64x64 tile per 256-thread WG, 8 waves (2M x 4N),
// double-buffered A tile staged with async-to-LDS.
// ---------------------------------------------------------------------------
__global__ __launch_bounds__(256) void xproj_gemm(
    const __bf16* __restrict__ xbf, const __bf16* __restrict__ w,
    const float* __restrict__ bias, float* __restrict__ xg) {
    __shared__ __bf16 As[2][64 * 32];       // 2 x 4 KB

    const int tid  = threadIdx.x;
    const int lane = tid & 31;
    const int wv   = tid >> 5;
    const int wc   = wv & 3;                // N position within tile (x16)
    const int wr   = wv >> 2;               // M band within tile (x32)

    const long mtile = (long)blockIdx.x * 64;
    const int  ntile = blockIdx.y * 64;

    const int r  = tid >> 2;                // A-tile row this thread copies
    const int c0 = (tid & 3) * 8;           // 8 bf16 = 16 B per thread

    v8f acc0 = {};
    v8f acc1 = {};

    // prologue: issue tile 0
    async_copy_b128(&As[0][r * 32 + c0], xbf + (mtile + r) * EMBED + c0);

    for (int it = 0; it < EMBED / 32; ++it) {
        const int p  = it & 1;
        const int kt = it * 32;
        if (it < EMBED / 32 - 1) {
            async_copy_b128(&As[p ^ 1][r * 32 + c0],
                            xbf + (mtile + r) * EMBED + kt + 32 + c0);
            WAIT_ASYNC(1);                  // my copy for buffer p has landed
        } else {
            WAIT_ASYNC(0);
        }
        __syncthreads();                    // everyone's copies have landed

        v16bf bf = load_frag_b(w + (size_t)(ntile + wc * 16) * EMBED + kt, EMBED, lane);
        v16bf a0 = load_frag_a(&As[p][(wr * 32 +  0) * 32], 32, lane);
        v16bf a1 = load_frag_a(&As[p][(wr * 32 + 16) * 32], 32, lane);
        acc0 = __builtin_amdgcn_wmma_f32_16x16x32_bf16(false, a0, false, bf,
                                                       (short)0, acc0, false, false);
        acc1 = __builtin_amdgcn_wmma_f32_16x16x32_bf16(false, a1, false, bf,
                                                       (short)0, acc1, false, false);
        __syncthreads();                    // buffer p free for reuse
    }

    // C/D layout: VGPR v -> M = v + 8*(lane>=16), N = lane&15
    const int n_e = ntile + wc * 16 + (lane & 15);
    const float bv = bias[n_e];
    const int hi8 = (lane >> 4) * 8;
    #pragma unroll
    for (int v = 0; v < 8; ++v) {
        long m0 = mtile + wr * 32 +  0 + v + hi8;
        long m1 = mtile + wr * 32 + 16 + v + hi8;
        xg[m0 * G4 + n_e] = acc0[v] + bv;
        xg[m1 * G4 + n_e] = acc1[v] + bv;
    }
}

// ---------------------------------------------------------------------------
// Kernel 2: recurrence.  One WG per 16-row batch slice (8 WGs; LSTM couples
// time and hidden, not batch -> no inter-WG sync).  Per step:
//   - async-copy xg[t, bs:bs+16, :] (128 KB contiguous) into `gates` LDS
//   - gates = act( xg_t + h_bf16 @ W_hh^T )   (WMMA, in-place in LDS)
//   - c/h elementwise update, h -> bf16 LDS + f32 global out
// LDS: 128 KB gates + 32 KB c + 16 KB h_bf16 = 176 KB of the 320 KB pool.
// ---------------------------------------------------------------------------
__global__ __launch_bounds__(256) void lstm_rec(
    const float* __restrict__ xg, const __bf16* __restrict__ whh,
    float* __restrict__ out) {
    extern __shared__ char smem[];
    float*  gates = (float*)smem;                                    // 16*2048 f32
    float*  cbuf  = (float*)(smem + 16 * G4 * 4);                    // 16*512  f32
    __bf16* hbf   = (__bf16*)(smem + 16 * G4 * 4 + 16 * HIDDEN * 4); // 16*512  bf16

    const int tid  = threadIdx.x;
    const int lane = tid & 31;
    const int wv   = tid >> 5;              // owns gate cols [wv*256, wv*256+256)
    const int bs   = blockIdx.x * 16;

    for (int i = tid; i < 16 * HIDDEN; i += 256) {
        cbuf[i] = 0.0f;
        hbf[i]  = (__bf16)0.0f;
    }
    __syncthreads();

    const int hi8 = (lane >> 4) * 8;
    const int nl  = lane & 15;

    for (int t = 0; t < TSTEPS; ++t) {
        // bulk async stage of this step's xg slice (contiguous 128 KB)
        {
            const char* src = (const char*)(xg + ((size_t)t * BATCH + bs) * G4);
            char* dst = (char*)gates;
            for (int ofs = tid * 16; ofs < 16 * G4 * 4; ofs += 256 * 16)
                async_copy_b128(dst + ofs, src + ofs);
        }

        // hoist all A fragments (h rows) while the copy is in flight
        v16bf afr[16];
        #pragma unroll
        for (int kk = 0; kk < 16; ++kk)
            afr[kk] = load_frag_a(hbf + kk * 32, HIDDEN, lane);

        WAIT_ASYNC(0);
        __syncthreads();                    // xg slice visible to all waves

        for (int nt = 0; nt < 16; ++nt) {
            const int n0 = wv * 256 + nt * 16;
            if (nt < 15)
                __builtin_prefetch((const void*)(whh + (size_t)(n0 + 16) * HIDDEN), 0, 1);

            v8f acc;
            #pragma unroll
            for (int v = 0; v < 8; ++v)
                acc[v] = gates[(v + hi8) * G4 + n0 + nl];

            #pragma unroll
            for (int kk = 0; kk < 16; ++kk) {
                v16bf bf = load_frag_b(whh + (size_t)n0 * HIDDEN + kk * 32, HIDDEN, lane);
                acc = __builtin_amdgcn_wmma_f32_16x16x32_bf16(false, afr[kk], false, bf,
                                                              (short)0, acc, false, false);
            }

            // quadrant: 0=i(sig) 1=f(sig) 2=g(tanh) 3=o(sig); write in-place
            const int q = n0 >> 9;
            #pragma unroll
            for (int v = 0; v < 8; ++v) {
                float xv = acc[v];
                float av = (q == 2) ? tanhf(xv) : sigmoidf_fast(xv);
                gates[(v + hi8) * G4 + n0 + nl] = av;
            }
        }
        __syncthreads();

        // elementwise c/h update + output store
        float* outt = out + ((size_t)t * BATCH + bs) * HIDDEN;
        for (int i = tid; i < 16 * HIDDEN; i += 256) {
            int m = i >> 9, j = i & (HIDDEN - 1);
            float iv = gates[m * G4 +        j];
            float fv = gates[m * G4 +  512 + j];
            float gv = gates[m * G4 + 1024 + j];
            float ov = gates[m * G4 + 1536 + j];
            float c  = fv * cbuf[i] + iv * gv;
            cbuf[i]  = c;
            float h  = ov * tanhf(c);
            hbf[i]   = (__bf16)h;
            outt[(size_t)m * HIDDEN + j] = h;
        }
        __syncthreads();                    // protects gates/hbf for next step
    }
}

// ---------------------------------------------------------------------------
extern "C" void kernel_launch(void* const* d_in, const int* in_sizes, int n_in,
                              void* d_out, int out_size, void* d_ws, size_t ws_size,
                              hipStream_t stream) {
    const float* x    = (const float*)d_in[0];
    const float* w_ih = (const float*)d_in[1];
    const float* w_hh = (const float*)d_in[2];
    const float* b_ih = (const float*)d_in[3];
    const float* b_hh = (const float*)d_in[4];
    float* out = (float*)d_out;

    char* ws = (char*)d_ws;
    __bf16* w_ih_bf = (__bf16*)(ws);                    //  2 MB @ 0
    __bf16* w_hh_bf = (__bf16*)(ws + (2ull << 20));     //  2 MB @ 2M
    float*  bias    = (float*)(ws + (4ull << 20));      //  8 KB @ 4M
    __bf16* xbf     = (__bf16*)(ws + (5ull << 20));     // 64 MB @ 5M
    float*  xg      = (float*)(ws + (69ull << 20));     // 512 MB @ 69M

    cvt_weights<<<(G4 * EMBED + 255) / 256, 256, 0, stream>>>(
        w_ih, w_hh, b_ih, b_hh, w_ih_bf, w_hh_bf, bias);

    cvt_x<<<(TSTEPS * BATCH * EMBED) / (256 * 8), 256, 0, stream>>>(x, xbf);

    dim3 g1(TSTEPS * BATCH / 64, G4 / 64);              // 1024 x 32 WGs
    xproj_gemm<<<g1, 256, 0, stream>>>(xbf, w_ih_bf, bias, xg);

    size_t lds = 16 * G4 * 4 + 16 * HIDDEN * 4 + 16 * HIDDEN * 2;  // 176 KB
    lstm_rec<<<BATCH / 16, 256, lds, stream>>>(xg, w_hh_bf, out);
}
